// LocalToMeAttentionBlock_33105607918095
// MI455X (gfx1250) — matrix operand order repaired and checked by reading, not compiled, not run on hardware
//
#include <hip/hip_runtime.h>
#include <hip/hip_bf16.h>
#include <math.h>

// ---------------- CDNA5 WMMA types ----------------
typedef __attribute__((ext_vector_type(16))) __bf16 v16bf;
typedef __attribute__((ext_vector_type(8)))  float  v8f;

// ---------------- problem constants ----------------
#define BB   2
#define SS   8192
#define DD   768
#define FF   2048
#define NSRC 4096
#define GG   64
#define HH   12
#define HDD  64
#define WW   16
#define NW   (SS / WW)        // 512
#define MTOT (BB * SS)        // 16384
#define RR   4

// ---------------- workspace arena (bytes) ----------------
static constexpr size_t SZ_ACT  = (size_t)MTOT * DD * 4;        // 50,331,648
static constexpr size_t SZ_T1   = (size_t)MTOT * FF * 4;        // 134,217,728
static constexpr size_t OFF_WB  = 0;                             // bf16 weights (14,254,080 B)
static constexpr size_t OFF_H   = 14254080;                      // h (f32), later o
static constexpr size_t OFF_Q   = OFF_H  + SZ_ACT;               // q, later x1
static constexpr size_t OFF_K   = OFF_Q  + SZ_ACT;               // k, later h2
static constexpr size_t OFF_V   = OFF_K  + SZ_ACT;               // v, later x2
static constexpr size_t OFF_T1  = OFF_V  + SZ_ACT;               // t1, later g
static constexpr size_t OFF_M   = OFF_T1 + SZ_T1;                // m (16384x64 f32)
static constexpr size_t OFF_Z   = OFF_M  + (size_t)MTOT * GG * 4;// z (16384 f32)

// bf16 weight sub-offsets (elements)
static constexpr size_t WE_DD = (size_t)DD * DD;   // 589824
static constexpr size_t WE_DF = (size_t)DD * FF;   // 1572864
static constexpr size_t WE_DG = (size_t)DD * GG;   // 49152

// =============== f32[K][N] -> bf16[N][K] transpose-convert ===============
__global__ __launch_bounds__(256)
void k_cvt_t(const float* __restrict__ s, __bf16* __restrict__ d, int K, int N) {
  __shared__ float tile[32][33];
  int kb = blockIdx.y * 32, nb = blockIdx.x * 32;
  int tx = threadIdx.x & 31, ty = threadIdx.x >> 5;   // 32 x 8
  for (int r = ty; r < 32; r += 8)
    tile[r][tx] = s[(size_t)(kb + r) * N + nb + tx];
  __syncthreads();
  for (int r = ty; r < 32; r += 8)
    d[(size_t)(nb + r) * K + kb + tx] = (__bf16)tile[tx][r];
}

// =============== RMSNorm (one row per block) ===============
__global__ __launch_bounds__(256)
void k_rmsnorm(const float* __restrict__ x, const float* __restrict__ w,
               float* __restrict__ out) {
  __shared__ float red[256];
  int row = blockIdx.x, t = threadIdx.x;
  const float* xr = x + (size_t)row * DD;
  float ss = 0.f;
  for (int c = t; c < DD; c += 256) { float v = xr[c]; ss += v * v; }
  red[t] = ss; __syncthreads();
  for (int off = 128; off > 0; off >>= 1) {
    if (t < off) red[t] += red[t + off];
    __syncthreads();
  }
  float rstd = rsqrtf(red[0] * (1.0f / DD) + 1e-6f);
  for (int c = t; c < DD; c += 256)
    out[(size_t)row * DD + c] = xr[c] * rstd * w[c];
}

// =============== Tiled bf16 WMMA GEMM: C[MxN] = A[MxK] @ W[KxN] ===============
// Weights pre-transposed: Wt is [N][K] bf16 so per-lane K is contiguous.
// MODE 0: C = acc      MODE 1: C = acc + Res      MODE 2: C = silu(Res)*acc
template<int MODE>
__global__ __launch_bounds__(256)
void k_gemm(const float* __restrict__ A, const __bf16* __restrict__ Wt,
            float* __restrict__ C, const float* __restrict__ Res,
            int M, int N, int K) {
  __shared__ __bf16 lds_a[64][40];    // 64 x 32 (+pad), row-major [m][k]
  __shared__ __bf16 lds_b[128][40];   // 128 x 32 (+pad), [n][k]
  int t   = threadIdx.x;
  int lid = t & 31, wid = t >> 5;
  int waveM = wid >> 2;               // 0..1  (32-row slab)
  int waveN = wid & 3;                // 0..3  (32-col slab)
  int m0 = blockIdx.y * 64;
  int n0 = blockIdx.x * 128;

  v8f acc[2][2];
  for (int i = 0; i < 2; ++i)
    for (int j = 0; j < 2; ++j)
      for (int r = 0; r < 8; ++r) acc[i][j][r] = 0.f;

  // staging assignments
  int arow = t >> 2;                  // 0..63
  int acol = (t & 3) * 8;             // 0,8,16,24
  int brow = t >> 1;                  // 0..127 (n within tile)
  int bks  = (t & 1) * 16;            // 0 or 16 (k segment)
  const float*  aptr = A  + (size_t)(m0 + arow) * K + acol;
  const __bf16* bptr = Wt + (size_t)(n0 + brow) * K + bks;
  bool bvalid = (n0 + brow) < N;

  for (int k0 = 0; k0 < K; k0 += 32) {
    // ---- stage A (f32 -> bf16), contiguous ----
    const float4* af = (const float4*)(aptr + k0);
    float4 a0 = af[0], a1 = af[1];
    if (k0 + 32 < K) {                 // WGP-scope prefetch of next K tiles
      __builtin_prefetch(aptr + k0 + 32, 0, 3);
      __builtin_prefetch(bptr + k0 + 32, 0, 3);
    }
    lds_a[arow][acol + 0] = (__bf16)a0.x;
    lds_a[arow][acol + 1] = (__bf16)a0.y;
    lds_a[arow][acol + 2] = (__bf16)a0.z;
    lds_a[arow][acol + 3] = (__bf16)a0.w;
    lds_a[arow][acol + 4] = (__bf16)a1.x;
    lds_a[arow][acol + 5] = (__bf16)a1.y;
    lds_a[arow][acol + 6] = (__bf16)a1.z;
    lds_a[arow][acol + 7] = (__bf16)a1.w;
    // ---- stage W tile, fully contiguous both sides (b128 in / b128 out) ----
    {
      uint4* ldst = (uint4*)&lds_b[brow][bks];
      if (bvalid) {
        const uint4* bf4 = (const uint4*)(bptr + k0);
        ldst[0] = bf4[0]; ldst[1] = bf4[1];
      } else {
        uint4 zz; zz.x = 0u; zz.y = 0u; zz.z = 0u; zz.w = 0u;
        ldst[0] = zz; ldst[1] = zz;
      }
    }
    __syncthreads();

    // ---- build fragments in gfx1250 WMMA layout ----
    union fr { v16bf v; unsigned u[8]; };
    fr afr[2], bfr[2];
    int kua = (lid >> 4) * 8;          // A: u32 offset (k-half 0 / 16)
    int kub = (lid >> 4) * 8;          // B: u32 offset (k-half 0 / 16)
    #pragma unroll
    for (int i = 0; i < 2; ++i) {
      const unsigned* ar = (const unsigned*)&lds_a[waveM * 32 + i * 16 + (lid & 15)][0];
      #pragma unroll
      for (int p = 0; p < 4; ++p) afr[i].u[p] = ar[kua + p];       // K = kb+0..7
      #pragma unroll
      for (int p = 4; p < 8; ++p) afr[i].u[p] = ar[kua + p];       // K = kb+8..15
    }
    #pragma unroll
    for (int j = 0; j < 2; ++j) {
      const unsigned* br = (const unsigned*)&lds_b[waveN * 32 + j * 16 + (lid & 15)][0];
      #pragma unroll
      for (int p = 0; p < 8; ++p) bfr[j].u[p] = br[kub + p];       // K = kb..kb+15
    }
    #pragma unroll
    for (int i = 0; i < 2; ++i)
      #pragma unroll
      for (int j = 0; j < 2; ++j)
        acc[i][j] = __builtin_amdgcn_wmma_f32_16x16x32_bf16(
            false, afr[i].v, false, bfr[j].v, (short)0, acc[i][j], false, false);
    __syncthreads();
  }

  // ---- epilogue ----
  #pragma unroll
  for (int i = 0; i < 2; ++i)
    #pragma unroll
    for (int j = 0; j < 2; ++j)
      #pragma unroll
      for (int r = 0; r < 8; ++r) {
        int row = m0 + waveM * 32 + i * 16 + r + ((lid >> 4) << 3);
        int col = n0 + waveN * 32 + j * 16 + (lid & 15);
        if (col < N) {
          size_t idx = (size_t)row * N + col;
          float v = acc[i][j][r];
          if (MODE == 1) v += Res[idx];
          if (MODE == 2) { float g = Res[idx]; v *= g / (1.f + __expf(-g)); }
          C[idx] = v;
        }
      }
}

// =============== RoPE in-place on q,k ===============
__global__ __launch_bounds__(256)
void k_rope(float* __restrict__ q, float* __restrict__ k,
            const int* __restrict__ pos, const float* __restrict__ fr) {
  int tok = blockIdx.x, t = threadIdx.x;
  int p = pos[tok];
  for (int idx = t; idx < HH * 32; idx += 256) {
    int h = idx >> 5, d = idx & 31;
    float a = fr[(size_t)p * 32 + d];
    float c = cosf(a), s = sinf(a);
    size_t base = (size_t)tok * DD + h * HDD + d;
    float q1 = q[base], q2 = q[base + 32];
    q[base] = q1 * c - q2 * s; q[base + 32] = q1 * s + q2 * c;
    float k1 = k[base], k2 = k[base + 32];
    k[base] = k1 * c - k2 * s; k[base + 32] = k1 * s + k2 * c;
  }
}

// =============== windowed attention: one (b, window, head) per block ===============
__global__ __launch_bounds__(256)
void k_attn(const float* __restrict__ q, const float* __restrict__ k,
            const float* __restrict__ v, float* __restrict__ o) {
  __shared__ float qs[16 * 64], ks[16 * 64], vs[16 * 64];
  __shared__ float sc[256];
  __shared__ float rmax[16], rsum[16];
  int t = threadIdx.x;
  int h = blockIdx.x % HH;
  int n = (blockIdx.x / HH) % NW;
  int b = blockIdx.x / (HH * NW);
  size_t tok0 = (size_t)b * SS + (size_t)n * WW;
  for (int idx = t; idx < 1024; idx += 256) {
    int i = idx >> 6, d = idx & 63;
    size_t g = (tok0 + i) * DD + h * HDD + d;
    qs[idx] = q[g]; ks[idx] = k[g]; vs[idx] = v[g];
  }
  __syncthreads();
  {
    int i = t >> 4, j = t & 15;
    float a = 0.f;
    #pragma unroll
    for (int d = 0; d < 64; ++d) a += qs[i * 64 + d] * ks[j * 64 + d];
    sc[t] = a * 0.125f;                 // 1/sqrt(64)
  }
  __syncthreads();
  if (t < 16) {
    float m = sc[t * 16];
    for (int j = 1; j < 16; ++j) m = fmaxf(m, sc[t * 16 + j]);
    rmax[t] = m;
  }
  __syncthreads();
  sc[t] = __expf(sc[t] - rmax[t >> 4]);
  __syncthreads();
  if (t < 16) {
    float s = 0.f;
    for (int j = 0; j < 16; ++j) s += sc[t * 16 + j];
    rsum[t] = s;
  }
  __syncthreads();
  for (int idx = t; idx < 1024; idx += 256) {
    int i = idx >> 6, d = idx & 63;
    float a = 0.f;
    #pragma unroll
    for (int j = 0; j < 16; ++j) a += sc[i * 16 + j] * vs[j * 64 + d];
    o[(tok0 + i) * DD + h * HDD + d] = a / rsum[i];
  }
}

// =============== normalize m rows (G=64) ===============
__global__ __launch_bounds__(64)
void k_mnorm(float* __restrict__ m) {
  __shared__ float red[64];
  int row = blockIdx.x, t = threadIdx.x;
  float v = m[(size_t)row * GG + t];
  red[t] = v * v; __syncthreads();
  for (int off = 32; off > 0; off >>= 1) {
    if (t < off) red[t] += red[t + off];
    __syncthreads();
  }
  m[(size_t)row * GG + t] = v * rsqrtf(red[0] + 1e-6f);
}

// =============== sizes = source.sum(-1) ===============
__global__ __launch_bounds__(256)
void k_rowsum(const float* __restrict__ src, float* __restrict__ z) {
  __shared__ float red[256];
  int row = blockIdx.x, t = threadIdx.x;
  const float* r = src + (size_t)row * NSRC;
  float s = 0.f;
  for (int c = t; c < NSRC; c += 256) s += r[c];
  red[t] = s; __syncthreads();
  for (int off = 128; off > 0; off >>= 1) {
    if (t < off) red[t] += red[t + off];
    __syncthreads();
  }
  if (t == 0) z[row] = red[0];
}

// =============== ToMe merge: one (b, window) per block ===============
__global__ __launch_bounds__(256)
void k_tome(const float* __restrict__ x2, const float* __restrict__ srcf,
            const int* __restrict__ pos, const float* __restrict__ m,
            const float* __restrict__ z,
            float* __restrict__ xout, float* __restrict__ sout,
            int* __restrict__ pout) {
  int blk = blockIdx.x;
  int n = blk % NW, b = blk / NW;
  int t = threadIdx.x;
  __shared__ float sma[8][64], smb[8][64];
  __shared__ float ssc[8][8];
  __shared__ float nm[8]; __shared__ int ni[8];
  __shared__ int s_src[4], s_unm[4], s_dst[4];
  __shared__ float s_zs[4], s_zb[8], s_zbn[8];
  size_t tok0 = (size_t)b * SS + (size_t)n * WW;
  for (int idx = t; idx < 512; idx += 256) {
    int i = idx >> 6, d = idx & 63;
    sma[i][d] = m[(tok0 + 2 * i) * GG + d];
    smb[i][d] = m[(tok0 + 2 * i + 1) * GG + d];
  }
  __syncthreads();
  if (t < 64) {
    int i = t >> 3, j = t & 7;
    float a = 0.f;
    #pragma unroll
    for (int d = 0; d < 64; ++d) a += sma[i][d] * smb[j][d];
    ssc[i][j] = a;
  }
  __syncthreads();
  if (t < 8) {
    float best = ssc[t][0]; int bj = 0;           // first-occurrence argmax
    for (int j = 1; j < 8; ++j) if (ssc[t][j] > best) { best = ssc[t][j]; bj = j; }
    nm[t] = best; ni[t] = bj;
    s_zb[t] = z[tok0 + 2 * t + 1];
  }
  __syncthreads();
  if (t == 0) {
    int order[8]; bool used[8] = {false, false, false, false, false, false, false, false};
    for (int s = 0; s < 8; ++s) {                 // stable descending argsort
      int bi = -1;
      for (int j = 0; j < 8; ++j)
        if (!used[j] && (bi < 0 || nm[j] > nm[bi])) bi = j;
      order[s] = bi; used[bi] = true;
    }
    for (int s = 0; s < RR; ++s) {
      s_src[s] = order[s];
      s_unm[s] = order[RR + s];
      s_dst[s] = ni[order[s]];
      s_zs[s]  = z[tok0 + 2 * order[s]];
    }
    for (int j = 0; j < 8; ++j) s_zbn[j] = s_zb[j];
    for (int s = 0; s < RR; ++s) s_zbn[s_dst[s]] += s_zs[s];
  }
  __syncthreads();
  size_t orow0 = ((size_t)b * NW + n) * 12;
  // x_out: 12 rows x 768
  for (int idx = t; idx < 12 * DD; idx += 256) {
    int c = idx / DD, d = idx - c * DD;
    float val;
    if (c < 4) {
      val = x2[(tok0 + 2 * s_unm[c]) * DD + d];
    } else {
      int j = c - 4;
      float a = x2[(tok0 + 2 * j + 1) * DD + d] * s_zb[j];
      #pragma unroll
      for (int s = 0; s < RR; ++s)
        if (s_dst[s] == j) a += x2[(tok0 + 2 * s_src[s]) * DD + d] * s_zs[s];
      val = a / s_zbn[j];
    }
    xout[(orow0 + c) * DD + d] = val;
  }
  // s_out: 12 rows x 4096
  for (int idx = t; idx < 12 * NSRC; idx += 256) {
    int c = idx / NSRC, g = idx - c * NSRC;
    float val;
    if (c < 4) {
      val = srcf[(tok0 + 2 * s_unm[c]) * NSRC + g];
    } else {
      int j = c - 4;
      float a = srcf[(tok0 + 2 * j + 1) * NSRC + g];
      #pragma unroll
      for (int s = 0; s < RR; ++s)
        if (s_dst[s] == j) a += srcf[(tok0 + 2 * s_src[s]) * NSRC + g];
      val = a;
    }
    sout[(orow0 + c) * NSRC + g] = val;
  }
  if (t < 12) {
    int c = t;
    int off = (c < 4) ? (2 * s_unm[c]) : (2 * (c - 4) + 1);
    pout[orow0 + c] = pos[tok0 + off];
  }
}

// ====================================================================
extern "C" void kernel_launch(void* const* d_in, const int* in_sizes, int n_in,
                              void* d_out, int out_size, void* d_ws, size_t ws_size,
                              hipStream_t stream) {
  const float* x    = (const float*)d_in[0];
  const float* srcf = (const float*)d_in[1];
  const int*   pos  = (const int*)d_in[2];
  const float* frq  = (const float*)d_in[3];
  // d_in[4] = r (==4, hardcoded)
  const float* n1w  = (const float*)d_in[5];
  const float* n2w  = (const float*)d_in[6];
  const float* wq   = (const float*)d_in[7];
  const float* wk   = (const float*)d_in[8];
  const float* wv   = (const float*)d_in[9];
  const float* wo   = (const float*)d_in[10];
  const float* w1   = (const float*)d_in[11];
  const float* w2   = (const float*)d_in[12];
  const float* w3   = (const float*)d_in[13];
  const float* wm   = (const float*)d_in[14];

  char* ws = (char*)d_ws;
  __bf16* wbq = (__bf16*)(ws + OFF_WB);
  __bf16* wbk = wbq + WE_DD;
  __bf16* wbv = wbk + WE_DD;
  __bf16* wbo = wbv + WE_DD;
  __bf16* wb1 = wbo + WE_DD;
  __bf16* wb2 = wb1 + WE_DF;
  __bf16* wb3 = wb2 + WE_DF;
  __bf16* wbm = wb3 + WE_DF;

  float* h  = (float*)(ws + OFF_H);   // also o (after qkv GEMMs)
  float* q  = (float*)(ws + OFF_Q);   // also x1
  float* k  = (float*)(ws + OFF_K);   // also h2
  float* v  = (float*)(ws + OFF_V);   // also x2
  float* t1 = (float*)(ws + OFF_T1);  // also g (silu-gated, in place)
  float* mm = (float*)(ws + OFF_M);
  float* z  = (float*)(ws + OFF_Z);
  float* o  = h;
  float* x1 = q;
  float* h2 = k;
  float* x2 = v;

  float* xout = (float*)d_out;
  float* sout = xout + (size_t)BB * (NW * 12) * DD;
  int*   pout = (int*)(sout + (size_t)BB * (NW * 12) * NSRC);

  // 1) weights -> bf16, transposed to [N][K]
  auto cvtT = [&](const float* s, __bf16* d, int K, int N) {
    dim3 g(N / 32, K / 32);
    k_cvt_t<<<g, 256, 0, stream>>>(s, d, K, N);
  };
  cvtT(wq, wbq, DD, DD); cvtT(wk, wbk, DD, DD);
  cvtT(wv, wbv, DD, DD); cvtT(wo, wbo, DD, DD);
  cvtT(w1, wb1, DD, FF); cvtT(w2, wb2, FF, DD);
  cvtT(w3, wb3, DD, FF); cvtT(wm, wbm, DD, GG);

  // 2) h = rmsnorm(x, norm1_w)
  k_rmsnorm<<<MTOT, 256, 0, stream>>>(x, n1w, h);

  // 3) q,k,v = h @ wq/wk/wv   (bf16 WMMA, f32 accum)
  dim3 gD(DD / 128, MTOT / 64);   // (6, 256)
  dim3 gF(FF / 128, MTOT / 64);   // (16, 256)
  dim3 gG(1, MTOT / 64);          // N=64 tile guard
  k_gemm<0><<<gD, 256, 0, stream>>>(h, wbq, q, nullptr, MTOT, DD, DD);
  k_gemm<0><<<gD, 256, 0, stream>>>(h, wbk, k, nullptr, MTOT, DD, DD);
  k_gemm<0><<<gD, 256, 0, stream>>>(h, wbv, v, nullptr, MTOT, DD, DD);

  // 4) RoPE (in place on q,k)
  k_rope<<<MTOT, 256, 0, stream>>>(q, k, pos, frq);

  // 5) windowed attention -> o (aliases h)
  k_attn<<<BB * NW * HH, 256, 0, stream>>>(q, k, v, o);

  // 6) x1 = x + o @ wo
  k_gemm<1><<<gD, 256, 0, stream>>>(o, wbo, x1, x, MTOT, DD, DD);

  // 7) h2 = rmsnorm(x1, norm2_w)
  k_rmsnorm<<<MTOT, 256, 0, stream>>>(x1, n2w, h2);

  // 8) t1 = h2 @ w1 ; g = silu(t1) * (h2 @ w3)  (in place over t1)
  k_gemm<0><<<gF, 256, 0, stream>>>(h2, wb1, t1, nullptr, MTOT, DD, FF);
  k_gemm<2><<<gF, 256, 0, stream>>>(h2, wb3, t1, t1, MTOT, DD, FF);

  // 9) x2 = x1 + g @ w2
  k_gemm<1><<<gD, 256, 0, stream>>>(t1, wb2, x2, x1, MTOT, FF, DD);

  // 10) m = normalize(x2 @ wm)
  k_gemm<0><<<gG, 256, 0, stream>>>(x2, wbm, mm, nullptr, MTOT, DD, GG);
  k_mnorm<<<MTOT, 64, 0, stream>>>(mm);

  // 11) z = source.sum(-1)
  k_rowsum<<<MTOT, 256, 0, stream>>>(srcf, z);

  // 12) token merge -> outputs
  k_tome<<<BB * NW, 256, 0, stream>>>(x2, srcf, pos, mm, z, xout, sout, pout);
}
